// CrossModalRouter_807453852053
// MI455X (gfx1250) — matrix operand order repaired and checked by reading, not compiled, not run on hardware
//
#include <hip/hip_runtime.h>

// ---------------- problem constants ----------------
#define T_    64
#define B_    4096
#define TD_   512
#define HD_   256
#define VD_   484
#define NPAD_ 512      // VD padded to multiple of 16
#define BT_   64       // batch rows per workgroup (halves L2 weight re-reads vs 32)
#define NBLK_ (B_ / BT_)
#define LN_EPS  1e-5f
#define COS_EPS 1e-8f

// ---------------- WMMA types ----------------
typedef __bf16 bf16_t;
typedef __attribute__((ext_vector_type(16))) bf16_t v16bf;
typedef __attribute__((ext_vector_type(8)))  float  v8f;

union FragU { uint4 u[2]; v16bf v; };

// native convert -> v_cvt_*bf16_f32 (RNE), much cheaper than integer RNE emulation
__device__ __forceinline__ bf16_t f2bf(float f) { return (bf16_t)f; }

// ---------------- LDS layout (bytes) ----------------
// Region X is a union: phase1 {As 64x40, Bs 256x40} halves  = 25600 B
//                      phase3 {Ws 512x40} halves             = 40960 B
//                      epilogue: 64-float reduction scratch
#define AS_OFF   0
#define BS_OFF   (64 * 40 * 2)                  // 5120
#define WS_OFF   0
#define RED_OFF  0
#define HF_OFF   40960                          // 64 x 264 floats = 67584 B
#define HB_OFF   (40960 + 67584)                // 64 x 272 halves = 34816 B  (@108544)
#define DOT_OFF  (40960 + 67584 + 34816)        // 64 floats (@143360)
#define NRM_OFF  (DOT_OFF + 256)                // 64 floats
#define SMEM_BYTES (NRM_OFF + 256)              // 143872 B -> 2 WGs / 320 KB WGP

// ============================================================
// Kernel 0: per-task centroid norms
// ============================================================
extern "C" __global__ void router_init(const float* __restrict__ cent,
                                       float* __restrict__ cnorm) {
  __shared__ float red[256];
  const int t = blockIdx.x, tid = threadIdx.x;
  float s = 0.f;
  for (int i = tid; i < VD_; i += 256) {
    float c = cent[(size_t)t * VD_ + i];
    s += c * c;
  }
  red[tid] = s;
  __syncthreads();
  for (int off = 128; off >= 1; off >>= 1) {
    if (tid < off) red[tid] += red[tid + off];
    __syncthreads();
  }
  if (tid == 0) cnorm[t] = fmaxf(sqrtf(red[0]), COS_EPS);
}

// ============================================================
// Kernel 1: fused  GEMM1 -> LN -> GELU -> GEMM2 -> cosine
// grid = (B/64, T), block = 256 threads (8 waves, wave32)
// ============================================================
extern "C" __global__ void __launch_bounds__(256)
router_main(const float* __restrict__ t_feat, const float* __restrict__ W1,
            const float* __restrict__ b1,     const float* __restrict__ gamma,
            const float* __restrict__ beta,   const float* __restrict__ W2,
            const float* __restrict__ b2,     const float* __restrict__ cent,
            const float* __restrict__ cnorm,  float* __restrict__ partials)
{
  extern __shared__ char smem[];
  bf16_t* As = (bf16_t*)(smem + AS_OFF);
  bf16_t* Bs = (bf16_t*)(smem + BS_OFF);
  bf16_t* Ws = (bf16_t*)(smem + WS_OFF);
  float*  redW = (float*)(smem + RED_OFF);
  float*  hF = (float*)(smem + HF_OFF);
  bf16_t* hB = (bf16_t*)(smem + HB_OFF);
  float*  dotL = (float*)(smem + DOT_OFF);
  float*  nrmL = (float*)(smem + NRM_OFF);

  const int tid  = threadIdx.x;
  const int lane = tid & 31;
  const int wave = tid >> 5;
  const int t    = blockIdx.y;
  const int b0   = blockIdx.x * BT_;

  const int mg = wave & 3;    // row group: rows [mg*16, mg*16+16)
  const int ng = wave >> 2;   // col group: phase1 128 cols, phase3 256 cols

  // ---------------- Phase 1: h = t_feat @ W1[t]^T  (64x256) ----------
  v8f acc1[8];
  const v8f vzero = {0.f,0.f,0.f,0.f,0.f,0.f,0.f,0.f};
#pragma unroll
  for (int j = 0; j < 8; ++j) acc1[j] = vzero;

  const float* W1t = W1 + (size_t)t * HD_ * TD_;

  for (int kc = 0; kc < TD_; kc += 32) {
    // stage A tile: 64 rows x 32 K  (fp32 -> bf16), 8 elems/thread
    {
      const int r = tid >> 2, c0 = (tid & 3) << 3;
      const float* src = t_feat + (size_t)(b0 + r) * TD_ + kc + c0;
      float4 f0 = *(const float4*)(src);
      float4 f1 = *(const float4*)(src + 4);
      bf16_t* p = As + r * 40 + c0;
      p[0]=f2bf(f0.x); p[1]=f2bf(f0.y); p[2]=f2bf(f0.z); p[3]=f2bf(f0.w);
      p[4]=f2bf(f1.x); p[5]=f2bf(f1.y); p[6]=f2bf(f1.z); p[7]=f2bf(f1.w);
    }
    // stage B tile: 256 rows x 32 K
    {
      const float* src = W1t + (size_t)tid * TD_ + kc;
      bf16_t* p = Bs + tid * 40;
#pragma unroll
      for (int c = 0; c < 32; c += 4) {
        float4 f = *(const float4*)(src + c);
        p[c]=f2bf(f.x); p[c+1]=f2bf(f.y); p[c+2]=f2bf(f.z); p[c+3]=f2bf(f.w);
      }
    }
    __syncthreads();

    // A fragment (16x32 bf16): lane<16 -> K {0..7,16..23}, lane>=16 -> +8
    FragU a;
    {
      const bf16_t* base = As + (mg * 16 + (lane & 15)) * 40 + ((lane >> 4) << 3);
      a.u[0] = *(const uint4*)(base);
      a.u[1] = *(const uint4*)(base + 16);
    }
#pragma unroll
    for (int j = 0; j < 8; ++j) {
      FragU b;   // B fragment (32x16): lane holds col N=lane%16, K {0..7,16..23}, lane>=16 -> +8
      const bf16_t* base = Bs + (ng * 128 + j * 16 + (lane & 15)) * 40 + ((lane >> 4) << 3);
      b.u[0] = *(const uint4*)(base);
      b.u[1] = *(const uint4*)(base + 16);
      acc1[j] = __builtin_amdgcn_wmma_f32_16x16x32_bf16(
          false, a.v, false, b.v, (short)0, acc1[j], false, false);
    }
    __syncthreads();
  }

  // spill h (pre-bias) to LDS fp32
#pragma unroll
  for (int j = 0; j < 8; ++j) {
    const int col = ng * 128 + j * 16 + (lane & 15);
    const int rb  = mg * 16 + ((lane >> 4) << 3);
#pragma unroll
    for (int r = 0; r < 8; ++r)
      hF[(rb + r) * 264 + col] = acc1[j][r];
  }
  __syncthreads();

  // ---------------- Phase 2: +b1, LayerNorm, exact GELU -> bf16 ------
  {
    const int row = tid >> 2;            // 0..63
    const int c0  = (tid & 3) * 64;      // 4 threads per row, 64 cols each
    const float* b1t = b1 + (size_t)t * HD_;
    float s = 0.f, ss = 0.f;
    for (int c = 0; c < 64; ++c) {
      float x = hF[row * 264 + c0 + c] + b1t[c0 + c];
      s += x; ss += x * x;
    }
#pragma unroll
    for (int m = 2; m >= 1; m >>= 1) {
      s  += __shfl_xor(s,  m, 4);
      ss += __shfl_xor(ss, m, 4);
    }
    const float mean = s * (1.f / HD_);
    const float var  = ss * (1.f / HD_) - mean * mean;
    const float rstd = rsqrtf(var + LN_EPS);
    const float* gt = gamma + (size_t)t * HD_;
    const float* bt = beta  + (size_t)t * HD_;
    for (int c = 0; c < 64; ++c) {
      float x  = hF[row * 264 + c0 + c] + b1t[c0 + c];
      float hn = (x - mean) * rstd * gt[c0 + c] + bt[c0 + c];
      float g  = 0.5f * hn * (1.f + erff(hn * 0.70710678118654752f));
      hB[row * 272 + c0 + c] = f2bf(g);
    }
  }
  if (tid < 64) { dotL[tid] = 0.f; nrmL[tid] = 0.f; }
  __syncthreads();

  // ---------------- Phase 3: v = h @ W2[t]^T, fused cosine -----------
  v8f acc2[16];
#pragma unroll
  for (int j = 0; j < 16; ++j) acc2[j] = vzero;
  const float* W2t = W2 + (size_t)t * VD_ * HD_;

  for (int kc = 0; kc < HD_; kc += 32) {
    // stage W2 tile: 512 (padded) rows x 32 K
#pragma unroll
    for (int h = 0; h < 2; ++h) {
      const int n = tid * 2 + h;
      bf16_t* p = Ws + n * 40;
      if (n < VD_) {
        const float* src = W2t + (size_t)n * HD_ + kc;
#pragma unroll
        for (int c = 0; c < 32; c += 4) {
          float4 f = *(const float4*)(src + c);
          p[c]=f2bf(f.x); p[c+1]=f2bf(f.y); p[c+2]=f2bf(f.z); p[c+3]=f2bf(f.w);
        }
      } else {
#pragma unroll
        for (int c = 0; c < 32; ++c) p[c] = f2bf(0.f);
      }
    }
    __syncthreads();

    FragU a;
    {
      const bf16_t* base = hB + (mg * 16 + (lane & 15)) * 272 + kc + ((lane >> 4) << 3);
      a.u[0] = *(const uint4*)(base);
      a.u[1] = *(const uint4*)(base + 16);
    }
#pragma unroll
    for (int j = 0; j < 16; ++j) {
      FragU b;
      const bf16_t* base = Ws + (ng * 256 + j * 16 + (lane & 15)) * 40 + ((lane >> 4) << 3);
      b.u[0] = *(const uint4*)(base);
      b.u[1] = *(const uint4*)(base + 16);
      acc2[j] = __builtin_amdgcn_wmma_f32_16x16x32_bf16(
          false, a.v, false, b.v, (short)0, acc2[j], false, false);
    }
    __syncthreads();
  }

  // fold in b2, reduce dot(v,c) and |v|^2 per row — v never leaves the WGP
  float pd[8], pn[8];
#pragma unroll
  for (int r = 0; r < 8; ++r) { pd[r] = 0.f; pn[r] = 0.f; }
  const float* ct  = cent + (size_t)t * VD_;
  const float* b2t = b2   + (size_t)t * VD_;
#pragma unroll
  for (int j = 0; j < 16; ++j) {
    const int col = ng * 256 + j * 16 + (lane & 15);
    float cv = 0.f, bv = 0.f;
    if (col < VD_) { cv = ct[col]; bv = b2t[col]; }
#pragma unroll
    for (int r = 0; r < 8; ++r) {
      float v = acc2[j][r] + bv;     // padded cols: acc==0, bv==0
      pd[r] += v * cv;
      pn[r] += v * v;
    }
  }
#pragma unroll
  for (int r = 0; r < 8; ++r) {
#pragma unroll
    for (int m = 8; m >= 1; m >>= 1) {
      pd[r] += __shfl_xor(pd[r], m, 16);
      pn[r] += __shfl_xor(pn[r], m, 16);
    }
  }
  if ((lane & 15) == 0) {
    const int rb = mg * 16 + ((lane >> 4) << 3);
#pragma unroll
    for (int r = 0; r < 8; ++r) {
      atomicAdd(&dotL[rb + r], pd[r]);
      atomicAdd(&nrmL[rb + r], pn[r]);
    }
  }
  __syncthreads();

  // per-row cosine distance, then workgroup reduction (64 rows span 2 waves)
  if (tid < 64) {
    const float cn = cnorm[t];
    const float vn = fmaxf(sqrtf(nrmL[tid]), COS_EPS);
    redW[tid] = 1.f - dotL[tid] / (vn * cn);
  }
  __syncthreads();
  if (tid < 32) {
    float part = redW[tid] + redW[tid + 32];
#pragma unroll
    for (int m = 16; m >= 1; m >>= 1) part += __shfl_xor(part, m, 32);
    if (tid == 0) partials[(size_t)t * gridDim.x + blockIdx.x] = part;
  }
}

// ============================================================
// Kernel 2: deterministic reduction -> distances[64], argmin
// ============================================================
extern "C" __global__ void router_final(const float* __restrict__ partials,
                                        float* __restrict__ out, int nblk) {
  const int t = threadIdx.x;
  if (t < T_) {
    float s = 0.f;
    for (int i = 0; i < nblk; ++i) s += partials[(size_t)t * nblk + i];
    out[t] = s * (1.f / B_);
  }
  __syncthreads();
  if (t == 0) {
    int best = 0; float bv = out[0];
    for (int i = 1; i < T_; ++i) {
      float v = out[i];
      if (v < bv) { bv = v; best = i; }
    }
    out[T_] = (float)best;
  }
}

// ============================================================
extern "C" void kernel_launch(void* const* d_in, const int* in_sizes, int n_in,
                              void* d_out, int out_size, void* d_ws, size_t ws_size,
                              hipStream_t stream) {
  (void)in_sizes; (void)n_in; (void)out_size; (void)ws_size;
  const float* t_feat = (const float*)d_in[0];
  const float* W1     = (const float*)d_in[1];
  const float* b1     = (const float*)d_in[2];
  const float* gamma  = (const float*)d_in[3];
  const float* beta   = (const float*)d_in[4];
  const float* W2     = (const float*)d_in[5];
  const float* b2     = (const float*)d_in[6];
  const float* cent   = (const float*)d_in[7];

  float* cnorm    = (float*)d_ws;        // 64 floats
  float* partials = cnorm + T_;          // 64 * 64 floats
  float* out      = (float*)d_out;       // 65 floats

  router_init<<<T_, 256, 0, stream>>>(cent, cnorm);
  dim3 grid(NBLK_, T_);
  router_main<<<grid, 256, SMEM_BYTES, stream>>>(t_feat, W1, b1, gamma, beta,
                                                 W2, b2, cent, cnorm, partials);
  router_final<<<1, 64, 0, stream>>>(partials, out, NBLK_);
}
